// MoELayer_74826920231221
// MI455X (gfx1250) — matrix operand order repaired and checked by reading, not compiled, not run on hardware
//
#include <hip/hip_runtime.h>
#include <math.h>

// MI455X (gfx1250, wave32) MoE layer: gather-routed fused expert MLP on
// v_wmma_f32_16x16x32_bf16. Compute-bound (309 GFLOP vs ~150MB HBM); all
// expert weights (37.7MB bf16) stay resident in the 192MB L2.

typedef __bf16 bf16;
typedef __bf16 v16bf __attribute__((ext_vector_type(16)));
typedef float  v8f   __attribute__((ext_vector_type(8)));

#define T_TOK 16384
#define HDIM  768
#define IDIM  3072
#define EEXP  4
#define ROWS  32
#define IHALF 1536

union FragU { v16bf v; uint4 q[2]; };

// A-matrix 16x32 bf16 fragment (per ISA 7.12.2): lane holds row M=lane&15;
// lanes 0-15 -> K {0..7,16..23}, lanes 16-31 -> K {8..15,24..31}.
__device__ __forceinline__ v16bf load_a_frag(const bf16* row, int kb, int lane) {
  const bf16* p = row + kb * 32 + ((lane >> 4) * 8);
  FragU u;
  u.q[0] = *(const uint4*)(p);
  u.q[1] = *(const uint4*)(p + 16);
  return u.v;
}

// B-matrix 32x16 bf16 fragment: lane holds column N=lane&15, K-major;
// lanes 0-15 -> K {0..7,16..23}, lanes 16-31 -> K {8..15,24..31}.
__device__ __forceinline__ v16bf load_b_frag(const bf16* colrow, int kb, int lane) {
  const bf16* p = colrow + kb * 32 + ((lane >> 4) * 8);
  FragU u;
  u.q[0] = *(const uint4*)(p);
  u.q[1] = *(const uint4*)(p + 16);
  return u.v;
}

// ---------------------------------------------------------------- init
__global__ __launch_bounds__(256) void init_kernel(float* __restrict__ out,
                                                   unsigned* __restrict__ cnt) {
  size_t i = (size_t)blockIdx.x * 256 + threadIdx.x;
  if (i < (size_t)T_TOK * HDIM / 4) {
    float4 z = {0.f, 0.f, 0.f, 0.f};
    ((float4*)out)[i] = z;
  }
  if (i < 4) cnt[i] = 0u;
}

// ------------------------------------------------- weight transpose->bf16
// w1[e]: [H,I] f32 -> w1t[e]: [I,H] bf16 ; w2[e]: [I,H] f32 -> w2t[e]: [H,I] bf16
__global__ __launch_bounds__(256) void transpose_kernel(const float* __restrict__ w1,
                                                        const float* __restrict__ w2,
                                                        bf16* __restrict__ w1t,
                                                        bf16* __restrict__ w2t) {
  __shared__ float tile[32][33];
  const int z = blockIdx.z;
  const float* in;
  bf16* outp;
  int R, C;
  if (z < 4) { in = w1 + (size_t)z * HDIM * IDIM; outp = w1t + (size_t)z * IDIM * HDIM; R = HDIM; C = IDIM; }
  else       { in = w2 + (size_t)(z - 4) * IDIM * HDIM; outp = w2t + (size_t)(z - 4) * HDIM * IDIM; R = IDIM; C = HDIM; }
  const int c0 = blockIdx.x * 32, r0 = blockIdx.y * 32;
  if (c0 >= C || r0 >= R) return;
  for (int k = 0; k < 32; k += 8) {
    int r = r0 + threadIdx.y + k, c = c0 + threadIdx.x;
    tile[threadIdx.y + k][threadIdx.x] = in[(size_t)r * C + c];
  }
  __syncthreads();
  for (int k = 0; k < 32; k += 8) {
    int oc = c0 + threadIdx.y + k;  // output row = original column
    int orr = r0 + threadIdx.x;     // output col = original row
    outp[(size_t)oc * R + orr] = (bf16)tile[threadIdx.x][threadIdx.y + k];
  }
}

// ---------------------------------------------------------------- router
// one wave32 per token: logits, softmax, top-2 -> gather lists + prob partials
__global__ __launch_bounds__(256) void router_kernel(const float* __restrict__ x,
                                                     const float* __restrict__ rw,
                                                     const float* __restrict__ rb,
                                                     unsigned* __restrict__ cnt,
                                                     unsigned* __restrict__ list,
                                                     float* __restrict__ partial) {
  __shared__ float pp[8][4];
  const int tid = threadIdx.x, wave = tid >> 5, lane = tid & 31;
  const int token = blockIdx.x * 8 + wave;
  const float* xr = x + (size_t)token * HDIM;
  const float4* rw4 = (const float4*)rw;  // [H][4] rows are float4
  float l0 = 0.f, l1 = 0.f, l2 = 0.f, l3 = 0.f;
  for (int c = lane; c < HDIM; c += 32) {
    float xv = xr[c];
    float4 w = rw4[c];
    l0 += xv * w.x; l1 += xv * w.y; l2 += xv * w.z; l3 += xv * w.w;
  }
#pragma unroll
  for (int off = 16; off; off >>= 1) {
    l0 += __shfl_xor(l0, off);
    l1 += __shfl_xor(l1, off);
    l2 += __shfl_xor(l2, off);
    l3 += __shfl_xor(l3, off);
  }
  float lg[4] = {l0 + rb[0], l1 + rb[1], l2 + rb[2], l3 + rb[3]};
  float mx = fmaxf(fmaxf(lg[0], lg[1]), fmaxf(lg[2], lg[3]));
  float p[4]; float s = 0.f;
#pragma unroll
  for (int e = 0; e < 4; ++e) { p[e] = expf(lg[e] - mx); s += p[e]; }
  float inv = 1.0f / s;
#pragma unroll
  for (int e = 0; e < 4; ++e) p[e] *= inv;

  if (lane == 0) {
    int i1 = 0;
#pragma unroll
    for (int e = 1; e < 4; ++e) if (p[e] > p[i1]) i1 = e;  // strict > : lowest idx on tie
    int i2 = (i1 == 0) ? 1 : 0;
#pragma unroll
    for (int e = 0; e < 4; ++e) if (e != i1 && p[e] > p[i2]) i2 = e;
    unsigned pos1 = atomicAdd(&cnt[i1], 1u); list[(size_t)i1 * T_TOK + pos1] = (unsigned)token;
    unsigned pos2 = atomicAdd(&cnt[i2], 1u); list[(size_t)i2 * T_TOK + pos2] = (unsigned)token;
    pp[wave][0] = p[0]; pp[wave][1] = p[1]; pp[wave][2] = p[2]; pp[wave][3] = p[3];
  }
  __syncthreads();
  if (tid < 4) {
    float s4 = 0.f;
#pragma unroll
    for (int w = 0; w < 8; ++w) s4 += pp[w][tid];
    partial[(size_t)blockIdx.x * 4 + tid] = s4;  // deterministic per-block partial
  }
}

// ------------------------------------------------------- load-balance loss
__global__ __launch_bounds__(256) void lb_kernel(const float* __restrict__ partial,
                                                 float* __restrict__ out) {
  __shared__ float s[256 * 4];
  const int tid = threadIdx.x;
  float a[4] = {0.f, 0.f, 0.f, 0.f};
  for (int b = tid; b < 2048; b += 256) {
    const float* pr = partial + (size_t)b * 4;
#pragma unroll
    for (int e = 0; e < 4; ++e) a[e] += pr[e];
  }
#pragma unroll
  for (int e = 0; e < 4; ++e) s[tid * 4 + e] = a[e];
  __syncthreads();
  for (int off = 128; off; off >>= 1) {
    if (tid < off)
#pragma unroll
      for (int e = 0; e < 4; ++e) s[tid * 4 + e] += s[(tid + off) * 4 + e];
    __syncthreads();
  }
  if (tid == 0) {
    float loss = 0.f;
#pragma unroll
    for (int e = 0; e < 4; ++e) {
      float imp = s[e] / (float)T_TOK;
      loss += 4.0f * imp * logf(imp);
    }
    out[(size_t)T_TOK * HDIM] = 0.01f * loss;
  }
}

// ------------------------------------------------------- fused expert MLP
// block = (expert, 32-token tile); 8 waves; fused GEMM1 -> GeLU(LDS) -> GEMM2.
__global__ __launch_bounds__(256) void moe_mlp_kernel(const float* __restrict__ x,
                                                      const bf16* __restrict__ w1t,
                                                      const float* __restrict__ b1,
                                                      const bf16* __restrict__ w2t,
                                                      const float* __restrict__ b2,
                                                      const unsigned* __restrict__ cnt,
                                                      const unsigned* __restrict__ list,
                                                      float* __restrict__ out) {
  __shared__ __align__(16) bf16 xs[ROWS * HDIM];    // 48 KB staged activations
  __shared__ __align__(16) bf16 hbuf[ROWS * IHALF]; // 96 KB GeLU intermediate (half of I)
  __shared__ int toks[ROWS];

  const int e = blockIdx.y;
  const unsigned ne = cnt[e];
  const unsigned tile0 = blockIdx.x * ROWS;
  if (tile0 >= ne) return;  // uniform exit: EXEC stays all-ones around WMMA

  const int tid = threadIdx.x;
  const int wave = tid >> 5;
  const int lane = tid & 31;
  const int laneN = lane & 15;

  if (tid < ROWS) {
    unsigned idx = tile0 + tid;
    toks[tid] = (idx < ne) ? (int)list[(size_t)e * T_TOK + idx] : -1;
  }
  __syncthreads();

  // stage x rows -> bf16 LDS (zero-fill invalid rows)
  for (int q = tid; q < ROWS * HDIM / 4; q += 256) {
    int r = q / (HDIM / 4);
    int c4 = q % (HDIM / 4);
    int tk = toks[r];
    float4 v = {0.f, 0.f, 0.f, 0.f};
    if (tk >= 0) v = *(const float4*)(x + (size_t)tk * HDIM + c4 * 4);
    union { bf16 h[4]; uint2 u; } pk;
    pk.h[0] = (bf16)v.x; pk.h[1] = (bf16)v.y; pk.h[2] = (bf16)v.z; pk.h[3] = (bf16)v.w;
    *(uint2*)(xs + r * HDIM + c4 * 4) = pk.u;
  }

  const bf16* w1e = w1t + (size_t)e * IDIM * HDIM;  // [I][H] K-contiguous
  const bf16* w2e = w2t + (size_t)e * HDIM * IDIM;  // [H][I] K-contiguous

  const v8f zero8 = {0.f, 0.f, 0.f, 0.f, 0.f, 0.f, 0.f, 0.f};
  v8f acc2[12];  // 12 persistent 16x16 output tiles per wave (2 rt x 6 ct)
#pragma unroll
  for (int t = 0; t < 12; ++t) acc2[t] = zero8;

  for (int half = 0; half < 2; ++half) {
    __syncthreads();  // xs ready / hbuf free
    // ---- Phase 1: hbuf = gelu(xs @ w1[:, half*1536 .. +1536) + b1)
#pragma unroll 1
    for (int t = 0; t < 24; ++t) {
      const int rt = t & 1;
      const int ct = wave * 12 + (t >> 1);  // [0,96)
      const int nloc = ct * 16 + laneN;     // [0,1536)
      const int nabs = half * IHALF + nloc; // [0,3072)
      const bf16* brow = w1e + (size_t)nabs * HDIM;
      const bf16* arow = xs + (rt * 16 + laneN) * HDIM;
      v8f acc = zero8;
#pragma unroll 4
      for (int kb = 0; kb < HDIM / 32; ++kb) {
        v16bf a = load_a_frag(arow, kb, lane);
        v16bf b = load_b_frag(brow, kb, lane);
        acc = __builtin_amdgcn_wmma_f32_16x16x32_bf16(false, a, false, b,
                                                      (short)0, acc, false, false);
      }
      const float bias = b1[e * IDIM + nabs];
#pragma unroll
      for (int v = 0; v < 8; ++v) {
        float xv = acc[v] + bias;
        float g = 0.5f * xv * (1.0f + erff(xv * 0.70710678118654752f));  // exact GeLU
        int m = rt * 16 + v + 8 * (lane >> 4);
        hbuf[m * IHALF + nloc] = (bf16)g;
      }
    }
    __syncthreads();
    // ---- Phase 2: acc2 += hbuf @ w2[half*1536 .. , :]
#pragma unroll 1
    for (int kb = 0; kb < IHALF / 32; ++kb) {
      v16bf a0 = load_a_frag(hbuf + (0 * 16 + laneN) * IHALF, kb, lane);
      v16bf a1 = load_a_frag(hbuf + (1 * 16 + laneN) * IHALF, kb, lane);
#pragma unroll
      for (int t = 0; t < 12; ++t) {
        const int rt = t & 1;
        const int ct = wave * 6 + (t >> 1);  // [0,48)
        const int n = ct * 16 + laneN;       // [0,768)
        const bf16* brow = w2e + (size_t)n * IDIM + half * IHALF;
        v16bf b = load_b_frag(brow, kb, lane);
        acc2[t] = __builtin_amdgcn_wmma_f32_16x16x32_bf16(false, (rt ? a1 : a0), false, b,
                                                          (short)0, acc2[t], false, false);
      }
    }
  }

  // ---- Epilogue: out[tok] += y + b2 (exactly one atomic per expert per elem;
  // 2 commuting f32 adds per output element -> deterministic result)
#pragma unroll
  for (int t = 0; t < 12; ++t) {
    const int rt = t & 1;
    const int ct = wave * 6 + (t >> 1);
    const int n = ct * 16 + laneN;
    const float bias = b2[e * HDIM + n];
#pragma unroll
    for (int v = 0; v < 8; ++v) {
      int m = rt * 16 + v + 8 * (lane >> 4);
      int tk = toks[m];
      if (tk >= 0) atomicAdd(out + (size_t)tk * HDIM + n, acc2[t][v] + bias);
    }
  }
}

// ---------------------------------------------------------------- launch
extern "C" void kernel_launch(void* const* d_in, const int* in_sizes, int n_in,
                              void* d_out, int out_size, void* d_ws, size_t ws_size,
                              hipStream_t stream) {
  const float* x  = (const float*)d_in[0];
  const float* w1 = (const float*)d_in[1];
  const float* b1 = (const float*)d_in[2];
  const float* w2 = (const float*)d_in[3];
  const float* b2 = (const float*)d_in[4];
  const float* rw = (const float*)d_in[5];
  const float* rb = (const float*)d_in[6];
  float* out = (float*)d_out;

  // workspace layout (~38.2 MB)
  char* ws = (char*)d_ws;
  unsigned* cnt   = (unsigned*)ws;                                   // 16 B
  unsigned* list  = (unsigned*)(ws + 256);                           // 4*T*4 = 256 KB
  float*   partial = (float*)(ws + 256 + 4 * T_TOK * 4);             // 2048*4*4 = 32 KB
  bf16*    w1t = (bf16*)(ws + 256 + 4 * T_TOK * 4 + 2048 * 4 * 4);   // 18.9 MB
  bf16*    w2t = w1t + (size_t)EEXP * IDIM * HDIM;                   // 18.9 MB

  init_kernel<<<(T_TOK * HDIM / 4 + 255) / 256, 256, 0, stream>>>(out, cnt);
  transpose_kernel<<<dim3(96, 96, 8), dim3(32, 8), 0, stream>>>(w1, w2, w1t, w2t);
  router_kernel<<<T_TOK / 8, 256, 0, stream>>>(x, rw, rb, cnt, list, partial);
  lb_kernel<<<1, 256, 0, stream>>>(partial, out);
  moe_mlp_kernel<<<dim3(T_TOK / ROWS, EEXP), 256, 0, stream>>>(x, w1t, b1, w2t, b2,
                                                               cnt, list, out);
}
